// EncoderBlock_29420525978034
// MI455X (gfx1250) — matrix operand order repaired and checked by reading, not compiled, not run on hardware
//
#include <hip/hip_runtime.h>
#include <hip/hip_bf16.h>
#include <hip/hip_fp16.h>

typedef __attribute__((ext_vector_type(16))) _Float16 v16h;
typedef __attribute__((ext_vector_type(8)))  float    v8f;

// ---- problem constants -------------------------------------------------
constexpr int Bq  = 4, Nn = 325, Tt = 48, Dd = 128, Hh = 8, HDd = 16;
constexpr int Mtok = Bq * Nn * Tt;          // 62400 tokens
constexpr float SCALE = 0.25f;              // HD^-0.5
constexpr float EPS   = 1e-5f;

union Frag16 { uint4 u[2]; v16h v; _Float16 h[16]; };

// CDNA5 async global->LDS copy (ASYNCcnt tracked), GVS mode:
// mem = SGPR64 + VGPR_i32_offset ; LDS byte addr in VGPR (generic addr[31:0])
#define ASYNC_G2L_B128(ldsoff, goff, base)                                   \
    asm volatile("global_load_async_to_lds_b128 %0, %1, %2"                  \
                 :: "v"(ldsoff), "v"(goff), "s"(base) : "memory")
#define WAIT_ASYNCCNT0() asm volatile("s_wait_asynccnt 0" ::: "memory")

// ---- fp32 -> fp16 convert ---------------------------------------------
__global__ void __launch_bounds__(256) cvt_f16(const float* __restrict__ s,
                                               _Float16* __restrict__ d, int n) {
    int i = blockIdx.x * 256 + threadIdx.x;
    if (i < n) d[i] = (_Float16)s[i];
}

// ---- LayerNorm: one wave per 128-wide row, optional (b,n,t)->(b,t,n) ---
__global__ void __launch_bounds__(256) ln_f16(const float* __restrict__ x,
                                              const float* __restrict__ g,
                                              const float* __restrict__ b,
                                              _Float16* __restrict__ out,
                                              int permute) {
    int wave = threadIdx.x >> 5, lane = threadIdx.x & 31;
    int row = blockIdx.x * 8 + wave;
    const float4 xv = *(const float4*)(x + (size_t)row * Dd + lane * 4);
    float s = xv.x + xv.y + xv.z + xv.w;
#pragma unroll
    for (int off = 16; off; off >>= 1) s += __shfl_xor(s, off, 32);
    float mean = s * (1.0f / 128.0f);
    float d0 = xv.x - mean, d1 = xv.y - mean, d2 = xv.z - mean, d3 = xv.w - mean;
    float vv = d0 * d0 + d1 * d1 + d2 * d2 + d3 * d3;
#pragma unroll
    for (int off = 16; off; off >>= 1) vv += __shfl_xor(vv, off, 32);
    float rstd = rsqrtf(vv * (1.0f / 128.0f) + EPS);
    int drow = row;
    if (permute) {
        int bb = row / (Nn * Tt);
        int rr = row % (Nn * Tt);
        int n = rr / Tt, t = rr % Tt;
        drow = (bb * Tt + t) * Nn + n;
    }
    int c = lane * 4;
    union { uint2 u; _Float16 h[4]; } o;
    o.h[0] = (_Float16)(d0 * rstd * g[c + 0] + b[c + 0]);
    o.h[1] = (_Float16)(d1 * rstd * g[c + 1] + b[c + 1]);
    o.h[2] = (_Float16)(d2 * rstd * g[c + 2] + b[c + 2]);
    o.h[3] = (_Float16)(d3 * rstd * g[c + 3] + b[c + 3]);
    *(uint2*)(out + (size_t)drow * Dd + c) = o.u;
}

// ---- WMMA GEMM: 64x128 tile, BK=32, double-buffered, 2x2 reg tile ------
// EPI 0: scatter fp16 into q/k/v [lead,h,L,16]
// EPI 1: outf[row,col]   = resid[row,col] + val
// EPI 2: outf[perm(row),col] = resid[perm(row),col] + val (spatial un-swap)
// EPI 3: outh[row,col] = fp16( gelu_exact(val) )
template <int EPI>
__global__ void __launch_bounds__(256)
gemm_wmma(const _Float16* __restrict__ A, const _Float16* __restrict__ W,
          const float* __restrict__ bias, int K, int Nc,
          float* __restrict__ outf, const float* __restrict__ resid,
          _Float16* __restrict__ qb, _Float16* __restrict__ kb,
          _Float16* __restrict__ vb, int L, _Float16* __restrict__ outh) {
    __shared__ _Float16 As[2][64 * 40];    // row-major [m][k], stride 40
    __shared__ _Float16 Bt[2][128 * 40];   // transposed [n][k], stride 40
    int tid = threadIdx.x;
    int lane = tid & 31, wave = tid >> 5;
    int wr = wave & 1, wc = wave >> 1;          // 2 x 4 wave grid -> 32x32 each
    int m0 = blockIdx.x * 64, n0 = blockIdx.y * 128;
    int mq = lane & 15, kh = lane >> 4;

    // per-thread staging coordinates
    int arow = tid >> 2, aseg = tid & 3;        // A: 64 rows x 4 b128 segs
    int brow = tid >> 3, bseg = tid & 7;        // B: 32 rows x 8 segs(16 halfs)
    unsigned ldsA0 = (unsigned)(size_t)(&As[0][arow * 40 + aseg * 8]);
    unsigned ldsA1 = (unsigned)(size_t)(&As[1][arow * 40 + aseg * 8]);
    unsigned long long abase = (unsigned long long)(const void*)A;
    int agoff = (int)(((size_t)(m0 + arow) * K + aseg * 8) * 2);
    const _Float16* wbase = W + (size_t)brow * Nc + n0 + bseg * 16;

    // prologue: stage tile 0
    ASYNC_G2L_B128(ldsA0, agoff, abase);
    {
        union { uint4 u; _Float16 h[8]; } t0, t1;
        t0.u = *(const uint4*)wbase;
        t1.u = *(const uint4*)(wbase + 8);
#pragma unroll
        for (int i = 0; i < 8; i++) Bt[0][(bseg * 16 + i) * 40 + brow] = t0.h[i];
#pragma unroll
        for (int i = 0; i < 8; i++) Bt[0][(bseg * 16 + 8 + i) * 40 + brow] = t1.h[i];
    }
    WAIT_ASYNCCNT0();
    __syncthreads();

    v8f c00 = {}, c01 = {}, c10 = {}, c11 = {};
    int nk = K >> 5;
    for (int i = 0; i < nk; i++) {
        int cur = i & 1, nxt = cur ^ 1;
        bool has = (i + 1) < nk;
        union { uint4 u; _Float16 h[8]; } nb0, nb1;
        if (has) {   // prefetch next tile: async A + B into VGPRs
            ASYNC_G2L_B128(cur ? ldsA0 : ldsA1, agoff + (i + 1) * 64, abase);
            const _Float16* wp = wbase + (size_t)(i + 1) * 32 * Nc;
            nb0.u = *(const uint4*)wp;
            nb1.u = *(const uint4*)(wp + 8);
        }
        // A fragments (16x32): e<8 -> k=8*kh+e ; e>=8 -> k=8+8*kh+e
        Frag16 fa0, fa1;
        {
            const _Float16* ap = &As[cur][(wr * 32 + mq) * 40];
            fa0.u[0] = *(const uint4*)(ap + kh * 8);
            fa0.u[1] = *(const uint4*)(ap + 16 + kh * 8);
            ap += 16 * 40;
            fa1.u[0] = *(const uint4*)(ap + kh * 8);
            fa1.u[1] = *(const uint4*)(ap + 16 + kh * 8);
        }
        // B fragments (32x16): e<8 -> k=8*kh+e ; e>=8 -> k=8+8*kh+e
        Frag16 fb0, fb1;
        {
            const _Float16* bp = &Bt[cur][(wc * 32 + mq) * 40 + kh * 8];
            fb0.u[0] = *(const uint4*)bp;  fb0.u[1] = *(const uint4*)(bp + 16);
            bp += 16 * 40;
            fb1.u[0] = *(const uint4*)bp;  fb1.u[1] = *(const uint4*)(bp + 16);
        }
        c00 = __builtin_amdgcn_wmma_f32_16x16x32_f16(false, fa0.v, false, fb0.v,
                                                     (short)0, c00, false, false);
        c01 = __builtin_amdgcn_wmma_f32_16x16x32_f16(false, fa0.v, false, fb1.v,
                                                     (short)0, c01, false, false);
        c10 = __builtin_amdgcn_wmma_f32_16x16x32_f16(false, fa1.v, false, fb0.v,
                                                     (short)0, c10, false, false);
        c11 = __builtin_amdgcn_wmma_f32_16x16x32_f16(false, fa1.v, false, fb1.v,
                                                     (short)0, c11, false, false);
        if (has) {   // sink transposed B stores for next tile
            _Float16* bt = &Bt[nxt][0];
#pragma unroll
            for (int j = 0; j < 8; j++) bt[(bseg * 16 + j) * 40 + brow] = nb0.h[j];
#pragma unroll
            for (int j = 0; j < 8; j++) bt[(bseg * 16 + 8 + j) * 40 + brow] = nb1.h[j];
        }
        WAIT_ASYNCCNT0();
        __syncthreads();
    }

    v8f accs[2][2] = {{c00, c01}, {c10, c11}};
#pragma unroll
    for (int ti = 0; ti < 2; ti++)
#pragma unroll
        for (int tj = 0; tj < 2; tj++) {
            v8f acc = accs[ti][tj];
#pragma unroll
            for (int r = 0; r < 8; r++) {
                int row = m0 + wr * 32 + ti * 16 + r + (kh << 3);
                int col = n0 + wc * 32 + tj * 16 + mq;
                float val = acc[r] + bias[col];
                if (EPI == 0) {
                    int lead = row / L, pos = row % L;
                    int which = col >> 7, hh = (col >> 4) & 7, hd = col & 15;
                    _Float16* dst = (which == 0) ? qb : ((which == 1) ? kb : vb);
                    dst[(((size_t)lead * Hh + hh) * L + pos) * HDd + hd] = (_Float16)val;
                } else if (EPI == 1) {
                    size_t idx = (size_t)row * Dd + col;
                    outf[idx] = resid[idx] + val;
                } else if (EPI == 2) {
                    int bb = row / (Tt * Nn);
                    int rr = row % (Tt * Nn);
                    int t = rr / Nn, n = rr % Nn;
                    size_t idx = ((size_t)((bb * Nn + n) * Tt + t)) * Dd + col;
                    outf[idx] = resid[idx] + val;
                } else {
                    float ge = 0.5f * val * (1.0f + erff(val * 0.70710678118654752f));
                    outh[(size_t)row * Nc + col] = (_Float16)ge;
                }
            }
        }
}

// ---- flash attention: one block per (lead, head), 32 keys per step -----
__global__ void __launch_bounds__(256)
attn_flash(const _Float16* __restrict__ Q, const _Float16* __restrict__ Kk,
           const _Float16* __restrict__ V, _Float16* __restrict__ out,
           const unsigned char* __restrict__ mask, int L, int leadDiv) {
    __shared__ _Float16 Ks[352 * 16];   // [key][hd]  (K^T frag: contiguous)
    __shared__ _Float16 Vt[16 * 352];   // [hd][key]  (V frag: contiguous in key)
    __shared__ _Float16 Ps[8 * 512];    // per-wave 16x32 P tile
    __shared__ _Float16 Zs[16];         // zero block for branchless K^T frag
    int lead = blockIdx.x >> 3;
    int h = blockIdx.x & 7;
    size_t base = ((size_t)lead * Hh + h) * (size_t)L * HDd;
    int tid = threadIdx.x, lane = tid & 31, wave = tid >> 5;
    int mq = lane & 15, kh = lane >> 4;
    int ntile = (L + 15) >> 4;
    int ntile2 = (ntile + 1) & ~1;                 // even # of key tiles
    int pad2 = ntile2 * 16;
    for (int i = tid; i < pad2 * 16; i += 256) {
        int r = i >> 4, c = i & 15;
        _Float16 kv = (_Float16)0, vv = (_Float16)0;
        if (r < L) { kv = Kk[base + i]; vv = V[base + i]; }
        Ks[i] = kv;
        Vt[c * 352 + r] = vv;           // transpose V on the way into LDS
    }
    if (tid < 16) Zs[tid] = (_Float16)0;
    __syncthreads();

    // per-lane key-validity bitset over key tiles (key = kt*16 + mq)
    unsigned okbits = 0;
    {
        int bb = mask ? (lead / leadDiv) : 0;
        for (int kt = 0; kt < ntile; kt++) {
            int key = kt * 16 + mq;
            bool ok = key < L;
            if (mask) ok = ok && (mask[(size_t)bb * Nn + key] != 0);
            okbits |= (ok ? 1u : 0u) << kt;
        }
    }

    for (int qt = wave; qt < ntile; qt += 8) {
        Frag16 fq;
#pragma unroll
        for (int e = 0; e < 16; e++) fq.h[e] = (_Float16)0;
        int qrow = qt * 16 + mq;
        if (qrow < L && kh == 0) {
#pragma unroll
            for (int e = 0; e < 16; e++)          // k = 16*kh + e (hd dim; kh==1 lanes zero)
                fq.h[e] = Q[base + (size_t)qrow * HDd + e];
        }
        float rmax[8], rsum[8]; v8f o = {};
#pragma unroll
        for (int r = 0; r < 8; r++) { rmax[r] = -1e30f; rsum[r] = 0.0f; }

        for (int kt2 = 0; kt2 < ntile2; kt2 += 2) {
            // two score tiles (branchless: kh==1 lanes read zeroed LDS block)
            const _Float16* kp0 = (kh == 0) ? &Ks[(kt2 * 16 + mq) * 16] : Zs;
            const _Float16* kp1 = (kh == 0) ? &Ks[((kt2 + 1) * 16 + mq) * 16] : Zs;
            Frag16 fk0, fk1;
            fk0.u[0] = *(const uint4*)kp0; fk0.u[1] = *(const uint4*)(kp0 + 8);
            fk1.u[0] = *(const uint4*)kp1; fk1.u[1] = *(const uint4*)(kp1 + 8);
            v8f zero = {};
            v8f s0 = __builtin_amdgcn_wmma_f32_16x16x32_f16(false, fq.v, false, fk0.v,
                                                            (short)0, zero, false, false);
            v8f s1 = __builtin_amdgcn_wmma_f32_16x16x32_f16(false, fq.v, false, fk1.v,
                                                            (short)0, zero, false, false);
            bool ok0 = (okbits >> kt2) & 1u;
            bool ok1 = (okbits >> (kt2 + 1)) & 1u;
            float p0[8], p1[8], tr[8];
#pragma unroll
            for (int r = 0; r < 8; r++) {
                float a = s0[r] * SCALE, b = s1[r] * SCALE;
                a = ok0 ? a : -1e30f;
                b = ok1 ? b : -1e30f;
                p0[r] = a; p1[r] = b;
                tr[r] = fmaxf(a, b);
            }
#pragma unroll
            for (int off = 1; off < 16; off <<= 1)
#pragma unroll
                for (int r = 0; r < 8; r++) tr[r] = fmaxf(tr[r], __shfl_xor(tr[r], off, 16));
            float ps[8];
#pragma unroll
            for (int r = 0; r < 8; r++) {
                float nm = fmaxf(rmax[r], tr[r]);
                float corr = __expf(rmax[r] - nm);
                rmax[r] = nm;
                p0[r] = __expf(p0[r] - nm);
                p1[r] = __expf(p1[r] - nm);
                ps[r] = p0[r] + p1[r];
                rsum[r] *= corr;
                o[r] *= corr;
            }
#pragma unroll
            for (int off = 1; off < 16; off <<= 1)
#pragma unroll
                for (int r = 0; r < 8; r++) ps[r] += __shfl_xor(ps[r], off, 16);
#pragma unroll
            for (int r = 0; r < 8; r++) rsum[r] += ps[r];

            // transpose 16x32 P (C layout -> A fragment) via per-wave LDS
            _Float16* pw = &Ps[wave * 512];
#pragma unroll
            for (int r = 0; r < 8; r++) {
                int rr = (r + (kh << 3)) * 32;
                pw[rr + mq]      = (_Float16)p0[r];
                pw[rr + 16 + mq] = (_Float16)p1[r];
            }
            Frag16 fp;
            {
                const _Float16* pp = pw + mq * 32;
                fp.u[0] = *(const uint4*)(pp + kh * 8);
                fp.u[1] = *(const uint4*)(pp + 16 + kh * 8);
            }
            Frag16 fv;                 // full B-frag of V: k = key local 0..31
            {
                const _Float16* vp = &Vt[mq * 352 + kt2 * 16 + kh * 8];
                fv.u[0] = *(const uint4*)vp;
                fv.u[1] = *(const uint4*)(vp + 16);
            }
            o = __builtin_amdgcn_wmma_f32_16x16x32_f16(false, fp.v, false, fv.v,
                                                       (short)0, o, false, false);
        }
#pragma unroll
        for (int r = 0; r < 8; r++) {
            int row = qt * 16 + r + (kh << 3);
            if (row < L)
                out[((size_t)lead * L + row) * Dd + h * HDd + mq] =
                    (_Float16)(o[r] / rsum[r]);
        }
    }
}

// ---- host orchestration ------------------------------------------------
extern "C" void kernel_launch(void* const* d_in, const int* in_sizes, int n_in,
                              void* d_out, int out_size, void* d_ws, size_t ws_size,
                              hipStream_t stream) {
    const float* x        = (const float*)d_in[0];
    const unsigned char* mask = (const unsigned char*)d_in[1];
    const float* ln1_s    = (const float*)d_in[2];
    const float* ln1_b    = (const float*)d_in[3];
    const float* t_qkv_w  = (const float*)d_in[4];
    const float* t_qkv_b  = (const float*)d_in[5];
    const float* t_proj_w = (const float*)d_in[6];
    const float* t_proj_b = (const float*)d_in[7];
    const float* ln2_s    = (const float*)d_in[8];
    const float* ln2_b    = (const float*)d_in[9];
    const float* s_qkv_w  = (const float*)d_in[10];
    const float* s_qkv_b  = (const float*)d_in[11];
    const float* s_proj_w = (const float*)d_in[12];
    const float* s_proj_b = (const float*)d_in[13];
    const float* ln3_s    = (const float*)d_in[14];
    const float* ln3_b    = (const float*)d_in[15];
    const float* ffn_w1   = (const float*)d_in[16];
    const float* ffn_b1   = (const float*)d_in[17];
    const float* ffn_w2   = (const float*)d_in[18];
    const float* ffn_b2   = (const float*)d_in[19];
    float* out = (float*)d_out;

    char* ws = (char*)d_ws;
    size_t off = 0;
    auto alloc = [&](size_t elems, size_t elsz) -> void* {
        void* p = ws + off;
        off += (elems * elsz + 255) & ~(size_t)255;
        return p;
    };
    _Float16* wqt = (_Float16*)alloc(128 * 384, 2);
    _Float16* wpt = (_Float16*)alloc(128 * 128, 2);
    _Float16* wqs = (_Float16*)alloc(128 * 384, 2);
    _Float16* wps = (_Float16*)alloc(128 * 128, 2);
    _Float16* w1f = (_Float16*)alloc(128 * 512, 2);
    _Float16* w2f = (_Float16*)alloc(512 * 128, 2);
    _Float16* xn  = (_Float16*)alloc((size_t)Mtok * 128, 2);
    _Float16* qb  = (_Float16*)alloc((size_t)Mtok * 128, 2);
    _Float16* kb  = (_Float16*)alloc((size_t)Mtok * 128, 2);
    _Float16* vb  = (_Float16*)alloc((size_t)Mtok * 128, 2);
    _Float16* ab  = (_Float16*)alloc((size_t)Mtok * 128, 2);
    float*    x1  = (float*)   alloc((size_t)Mtok * 128, 4);
    _Float16* h1  = qb;  // FFN hidden (M x 512 f16) aliases q/k/v/attn region

    auto cvt = [&](const float* s, _Float16* d, int n) {
        cvt_f16<<<(n + 255) / 256, 256, 0, stream>>>(s, d, n);
    };
    cvt(t_qkv_w, wqt, 128 * 384);
    cvt(t_proj_w, wpt, 128 * 128);
    cvt(s_qkv_w, wqs, 128 * 384);
    cvt(s_proj_w, wps, 128 * 128);
    cvt(ffn_w1, w1f, 128 * 512);
    cvt(ffn_w2, w2f, 512 * 128);

    dim3 blk(256);
    int lnGrid = Mtok / 8;                 // 7800
    int gm = Mtok / 64;                    // 975

    // ---- temporal attention branch ----
    ln_f16<<<lnGrid, blk, 0, stream>>>(x, ln1_s, ln1_b, xn, 0);
    gemm_wmma<0><<<dim3(gm, 3), blk, 0, stream>>>(xn, wqt, t_qkv_b, 128, 384,
                                                  nullptr, nullptr, qb, kb, vb, Tt, nullptr);
    attn_flash<<<Bq * Nn * Hh, blk, 0, stream>>>(qb, kb, vb, ab, nullptr, Tt, 1);
    gemm_wmma<1><<<dim3(gm, 1), blk, 0, stream>>>(ab, wpt, t_proj_b, 128, 128,
                                                  x1, x, nullptr, nullptr, nullptr, 1, nullptr);

    // ---- spatial attention branch (swapaxes fused into LN + epilogue) ----
    ln_f16<<<lnGrid, blk, 0, stream>>>(x1, ln2_s, ln2_b, xn, 1);
    gemm_wmma<0><<<dim3(gm, 3), blk, 0, stream>>>(xn, wqs, s_qkv_b, 128, 384,
                                                  nullptr, nullptr, qb, kb, vb, Nn, nullptr);
    attn_flash<<<Bq * Tt * Hh, blk, 0, stream>>>(qb, kb, vb, ab, mask, Nn, Tt);
    gemm_wmma<2><<<dim3(gm, 1), blk, 0, stream>>>(ab, wps, s_proj_b, 128, 128,
                                                  out, x1, nullptr, nullptr, nullptr, 1, nullptr);

    // ---- FFN ----
    ln_f16<<<lnGrid, blk, 0, stream>>>(out, ln3_s, ln3_b, xn, 0);
    gemm_wmma<3><<<dim3(gm, 4), blk, 0, stream>>>(xn, w1f, ffn_b1, 128, 512,
                                                  nullptr, nullptr, nullptr, nullptr, nullptr, 1, h1);
    gemm_wmma<1><<<dim3(gm, 1), blk, 0, stream>>>(h1, w2f, ffn_b2, 512, 128,
                                                  out, out, nullptr, nullptr, nullptr, 1, nullptr);
    (void)in_sizes; (void)n_in; (void)out_size; (void)ws_size;
}
